// get_model_50002009260619
// MI455X (gfx1250) — compile-verified, matrix operand order checked by reading, not run on hardware
//
#include <hip/hip_runtime.h>

typedef __attribute__((ext_vector_type(16))) _Float16 v16h;
typedef __attribute__((ext_vector_type(8)))  _Float16 v8h;
typedef __attribute__((ext_vector_type(8)))  float    v8f;

// ---------------------------------------------------------------------------
// WMMA GEMM:  Y[M,O] = X[M,Kp] * Wp[Op,Kp]^T + bias   (f16 in, f32 out)
// X, Wp are pre-padded f16 (Kp % 32 == 0, Op % 64 == 0, zero filled).
// M % 128 == 0 (guaranteed by caller). Block = 256 threads = 8 wave32s.
// Block tile 128(M) x 64(N): wave w owns rows [16w,16w+16), 4 accumulators.
// A fragments load straight from global (rows are wave-private);
// B tile is shared by all waves -> staged once in LDS.
// ---------------------------------------------------------------------------
#define GEMM_BM 128
#define GEMM_BN 64
#define GEMM_BK 32
#define GEMM_T  256

__global__ __launch_bounds__(GEMM_T)
void gemm_bias_wmma(const _Float16* __restrict__ X, int ldx,
                    const _Float16* __restrict__ Wp,
                    const float* __restrict__ bias, float* __restrict__ Y,
                    int M, int O) {
  __shared__ _Float16 Bsh[GEMM_BN][GEMM_BK];
  const int tid  = threadIdx.x;
  const int wave = tid >> 5;
  const int lane = tid & 31;
  const int half = lane >> 4;
  const int l16  = lane & 15;
  const int m0 = blockIdx.x * GEMM_BM;
  const int n0 = blockIdx.y * GEMM_BN;
  const int nk = ldx / GEMM_BK;              // ldx == Kp, multiple of 32
  v8f acc[4] = {{}, {}, {}, {}};

  const int arow = m0 + wave * 16 + l16;     // < M always (M % 128 == 0)
  const _Float16* aptr = X + (long)arow * ldx;
  const int srow = tid >> 1;                 // B staging row (threads 0..127)
  const int scol = (tid & 1) * 16;

  for (int kb = 0; kb < nk; ++kb) {
    const int k0 = kb * GEMM_BK;
    // ---- stage B tile (64 x 32) : guard-free b128 loads (Wp padded)
    if (tid < 128) {
      const v8h* src = (const v8h*)(Wp + (long)(n0 + srow) * ldx + k0 + scol);
      *(v8h*)&Bsh[srow][scol + 0] = src[0];  // ds_store_b128
      *(v8h*)&Bsh[srow][scol + 8] = src[1];
    }
    // ---- A fragment direct from global, ISA 16-bit 16x32 layout:
    //  lanes 0-15: K 0..7 / 16..23 ; lanes 16-31: K 8..15 / 24..31
    v8h a_lo = *(const v8h*)(aptr + k0 + 8 * half);
    v8h a_hi = *(const v8h*)(aptr + k0 + 16 + 8 * half);
    v16h a = __builtin_shufflevector(a_lo, a_hi, 0, 1, 2, 3, 4, 5, 6, 7,
                                     8, 9, 10, 11, 12, 13, 14, 15);
    if (kb + 1 < nk) __builtin_prefetch(aptr + k0 + GEMM_BK, 0, 1);
    __syncthreads();
#pragma unroll
    for (int ns = 0; ns < 4; ++ns) {
      const int nrow = ns * 16 + l16;
      v8h b_lo = *(const v8h*)&Bsh[nrow][8 * half];
      v8h b_hi = *(const v8h*)&Bsh[nrow][16 + 8 * half];
      v16h b = __builtin_shufflevector(b_lo, b_hi, 0, 1, 2, 3, 4, 5, 6, 7,
                                       8, 9, 10, 11, 12, 13, 14, 15);
      acc[ns] = __builtin_amdgcn_wmma_f32_16x16x32_f16(
          /*neg_a=*/false, a, /*neg_b=*/false, b,
          /*c_mod=*/(short)0, acc[ns], /*reuse_a=*/false, /*reuse_b=*/false);
    }
    __syncthreads();
  }

  // ---- epilogue: C layout VGPR j: lanes0-15 -> M=j, lanes16-31 -> M=8+j
#pragma unroll
  for (int ns = 0; ns < 4; ++ns) {
    const int col = n0 + ns * 16 + l16;
#pragma unroll
    for (int j = 0; j < 8; ++j) {
      int row = m0 + wave * 16 + j + 8 * half;
      if (col < O) Y[(long)row * O + col] = acc[ns][j] + bias[col];
    }
  }
}

// ---------------------------------------------------------------------------
// Weight repack: W (O x K f32, row-major) -> Wp (Op x Kp f16, zero padded)
// ---------------------------------------------------------------------------
__global__ void pack_w(const float* __restrict__ W, _Float16* __restrict__ Wp,
                       int O, int K, int Kp, int total) {
  int i = blockIdx.x * blockDim.x + threadIdx.x;
  if (i >= total) return;                    // total = Op * Kp
  int o = i / Kp, k = i % Kp;
  Wp[i] = (_Float16)((o < O && k < K) ? W[(long)o * K + k] : 0.f);
}

// ---------------------------------------------------------------------------
// BatchNorm (population stats over rows, per channel) + ReLU, two pass.
// Apply writes f32 in place AND a zero-padded f16 copy for the next GEMM.
// ---------------------------------------------------------------------------
__global__ void bn_reduce(const float* __restrict__ Y, int M, int O,
                          float* __restrict__ sum, float* __restrict__ sumsq) {
  int c = blockIdx.x;                        // one block per channel
  float s = 0.f, s2 = 0.f;
  for (int r = threadIdx.x; r < M; r += 256) {
    float v = Y[(long)r * O + c];
    s += v; s2 += v * v;
  }
  __shared__ float sh[256], sh2[256];
  sh[threadIdx.x] = s; sh2[threadIdx.x] = s2;
  __syncthreads();
  for (int off = 128; off > 0; off >>= 1) {
    if (threadIdx.x < off) {
      sh[threadIdx.x]  += sh[threadIdx.x + off];
      sh2[threadIdx.x] += sh2[threadIdx.x + off];
    }
    __syncthreads();
  }
  if (threadIdx.x == 0) { sum[c] = sh[0]; sumsq[c] = sh2[0]; }
}

__global__ void bn_apply_relu(float* __restrict__ Y, int M, int O,
                              const float* __restrict__ sum, const float* __restrict__ sumsq,
                              const float* __restrict__ gamma, const float* __restrict__ beta,
                              _Float16* __restrict__ Xh, int Kp) {
  long i = (long)blockIdx.x * blockDim.x + threadIdx.x;
  long total = (long)M * Kp;
  if (i >= total) return;
  int c  = (int)(i % Kp);
  long r = i / Kp;
  if (c >= O) { Xh[i] = (_Float16)0.f; return; }   // zero pad
  long yi = r * O + c;
  float inv = 1.f / (float)M;
  float m   = sum[c] * inv;
  float var = sumsq[c] * inv - m * m;
  var = var > 0.f ? var : 0.f;
  float v = (Y[yi] - m) * rsqrtf(var + 1e-5f) * gamma[c] + beta[c];
  v = v > 0.f ? v : 0.f;
  Y[yi] = v;
  Xh[i] = (_Float16)v;
}

// ---------------------------------------------------------------------------
// Data prep: xyz (B,9,N) -> pts (B,N,9), coords (B,N,3)
// ---------------------------------------------------------------------------
__global__ void prep_pts(const float* __restrict__ xyz, float* __restrict__ pts,
                         float* __restrict__ coords, int Nn, int total) {
  int i = blockIdx.x * blockDim.x + threadIdx.x;
  if (i >= total) return;
  int c = i % 9;
  int n = (i / 9) % Nn;
  int b = i / (9 * Nn);
  float v = xyz[((long)b * 9 + c) * Nn + n];
  pts[i] = v;
  if (c < 3) coords[((long)b * Nn + n) * 3 + c] = v;
}

// ---------------------------------------------------------------------------
// Farthest point sampling: one block per batch, iterative block-wide argmax
// ---------------------------------------------------------------------------
#define FPS_T 256
__global__ __launch_bounds__(FPS_T)
void fps_kernel(const float* __restrict__ coords, int Nn, int npoint,
                int* __restrict__ fps_idx, float* __restrict__ new_xyz,
                float* __restrict__ dist) {
  int b = blockIdx.x;
  const float* cb = coords + (long)b * Nn * 3;
  float* db = dist + (long)b * Nn;
  for (int n = threadIdx.x; n < Nn; n += FPS_T) db[n] = 1e10f;
  __shared__ float sval[FPS_T];
  __shared__ int   sidx[FPS_T];
  __shared__ int   sfar;
  if (threadIdx.x == 0) sfar = 0;
  __syncthreads();
  for (int it = 0; it < npoint; ++it) {
    int far = sfar;
    if (threadIdx.x == 0) {
      fps_idx[b * npoint + it] = far;
      new_xyz[((long)b * npoint + it) * 3 + 0] = cb[far * 3 + 0];
      new_xyz[((long)b * npoint + it) * 3 + 1] = cb[far * 3 + 1];
      new_xyz[((long)b * npoint + it) * 3 + 2] = cb[far * 3 + 2];
    }
    float fx = cb[far * 3], fy = cb[far * 3 + 1], fz = cb[far * 3 + 2];
    float best = -1.f; int bi = 0x7fffffff;
    for (int n = threadIdx.x; n < Nn; n += FPS_T) {
      float dx = cb[n * 3] - fx, dy = cb[n * 3 + 1] - fy, dz = cb[n * 3 + 2] - fz;
      float d2 = dx * dx + dy * dy + dz * dz;
      float old = db[n];
      float nd = old < d2 ? old : d2;
      db[n] = nd;
      if (nd > best) { best = nd; bi = n; }   // ascending n -> first max kept
    }
    sval[threadIdx.x] = best; sidx[threadIdx.x] = bi;
    __syncthreads();
    for (int off = FPS_T / 2; off > 0; off >>= 1) {
      if (threadIdx.x < off) {
        float v2 = sval[threadIdx.x + off]; int i2 = sidx[threadIdx.x + off];
        if (v2 > sval[threadIdx.x] ||
            (v2 == sval[threadIdx.x] && i2 < sidx[threadIdx.x])) {
          sval[threadIdx.x] = v2; sidx[threadIdx.x] = i2;
        }
      }
      __syncthreads();
    }
    if (threadIdx.x == 0) sfar = sidx[0];
    __syncthreads();
  }
}

// ---------------------------------------------------------------------------
// Ball query: first Ksamp in-radius indices in ascending order, pad w/ first
// ---------------------------------------------------------------------------
__global__ void ball_query(const float* __restrict__ coords, const float* __restrict__ nxyz,
                           int Nn, int S, int Ksamp, float r2,
                           int* __restrict__ gidx, int total) {
  int t = blockIdx.x * blockDim.x + threadIdx.x;
  if (t >= total) return;                 // total = B*S
  int b = t / S;
  const float* cb = coords + (long)b * Nn * 3;
  float qx = nxyz[t * 3 + 0], qy = nxyz[t * 3 + 1], qz = nxyz[t * 3 + 2];
  int* out = gidx + (long)t * Ksamp;
  int cnt = 0;
  for (int n = 0; n < Nn && cnt < Ksamp; ++n) {
    float dx = cb[n * 3] - qx, dy = cb[n * 3 + 1] - qy, dz = cb[n * 3 + 2] - qz;
    float d2 = dx * dx + dy * dy + dz * dz;
    if (d2 <= r2) { out[cnt++] = n; }
  }
  int first = (cnt > 0) ? out[0] : 0;
  for (int k = cnt; k < Ksamp; ++k) out[k] = first;
}

// ---------------------------------------------------------------------------
// Gather grouped points -> padded f16 GEMM input:
// Xh[b,s,k,:] = [feats[b,gi,:], coords[b,gi]-new_xyz[b,s], 0-pad]
// ---------------------------------------------------------------------------
__global__ void gather_kernel(const float* __restrict__ feats, int Cf,
                              const float* __restrict__ coords, const float* __restrict__ nxyz,
                              const int* __restrict__ gidx, int Nn, int S, int Ksamp,
                              _Float16* __restrict__ Xh, int Cinp, int total) {
  int t = blockIdx.x * blockDim.x + threadIdx.x;
  if (t >= total) return;                 // total = B*S*Ksamp
  int sq = t / Ksamp;                     // b*S + s
  int b  = sq / S;
  int gi = gidx[t];
  const float* fsrc = feats + ((long)b * Nn + gi) * Cf;
  _Float16* dst = Xh + (long)t * Cinp;
  for (int c = 0; c < Cf; ++c) dst[c] = (_Float16)fsrc[c];
  const float* csrc = coords + ((long)b * Nn + gi) * 3;
  dst[Cf + 0] = (_Float16)(csrc[0] - nxyz[(long)sq * 3 + 0]);
  dst[Cf + 1] = (_Float16)(csrc[1] - nxyz[(long)sq * 3 + 1]);
  dst[Cf + 2] = (_Float16)(csrc[2] - nxyz[(long)sq * 3 + 2]);
  for (int c = Cf + 3; c < Cinp; ++c) dst[c] = (_Float16)0.f;
}

// ---------------------------------------------------------------------------
// Max pool over neighbors -> concat slot [coff, coff+O) of feats_next
// ---------------------------------------------------------------------------
__global__ void maxpool_kernel(const float* __restrict__ Y, int S, int Ksamp, int O,
                               float* __restrict__ outF, int Ctot, int coff, int total) {
  int t = blockIdx.x * blockDim.x + threadIdx.x;
  if (t >= total) return;                 // total = B*S*O
  int o  = t % O;
  int sq = t / O;                         // b*S + s
  const float* yp = Y + ((long)sq * Ksamp) * O + o;
  float m = yp[0];
  for (int k = 1; k < Ksamp; ++k) { float v = yp[(long)k * O]; m = v > m ? v : m; }
  outF[(long)sq * Ctot + coff + o] = m;
}

// ---------------------------------------------------------------------------
// l3h = concat([coords3, feats3], -1) -> padded f16 (1024 x 544)
// ---------------------------------------------------------------------------
__global__ void concat_l3(const float* __restrict__ c3, const float* __restrict__ f3,
                          _Float16* __restrict__ l3h, int total) {
  int t = blockIdx.x * blockDim.x + threadIdx.x;
  if (t >= total) return;                 // total = 1024*544
  int r = t / 544, c = t % 544;
  float v = (c < 3) ? c3[r * 3 + c] : ((c < 515) ? f3[r * 512 + (c - 3)] : 0.f);
  l3h[t] = (_Float16)v;
}

// ---------------------------------------------------------------------------
// Final outputs: [(coords+disp)^T (B,3,256)] [dist^T (B,1,256)] [coords^T]
// ---------------------------------------------------------------------------
__global__ void final_out(const float* __restrict__ c3, const float* __restrict__ disp,
                          const float* __restrict__ dist, float* __restrict__ out) {
  int t = blockIdx.x * blockDim.x + threadIdx.x;
  if (t >= 7168) return;
  if (t < 3072) {
    int b = t / 768, d = (t / 256) % 3, s = t % 256;
    int r = (b * 256 + s) * 3 + d;
    out[t] = c3[r] + disp[r];
  } else if (t < 4096) {
    int u = t - 3072;
    out[t] = dist[u];                     // (b*256+s), O=1
  } else {
    int u = t - 4096;
    int b = u / 768, d = (u / 256) % 3, s = u % 256;
    out[t] = c3[(b * 256 + s) * 3 + d];
  }
}

// ---------------------------------------------------------------------------
// Host orchestration
// ---------------------------------------------------------------------------
extern "C" void kernel_launch(void* const* d_in, const int* in_sizes, int n_in,
                              void* d_out, int out_size, void* d_ws, size_t ws_size,
                              hipStream_t stream) {
  (void)in_sizes; (void)n_in; (void)out_size; (void)ws_size;
  const int B = 4, N0 = 16384;

  // ---- unpack params (pytree order: disp, dist, sa; each conv = W,b,g,beta)
  int pi = 1;
  auto P = [&]() { return (const float*)d_in[pi++]; };
  struct Conv { const float *W, *b, *g, *be; };
  Conv disp[2], dist[2], sa[3][2][3];
  for (int i = 0; i < 2; i++) { disp[i].W = P(); disp[i].b = P(); disp[i].g = P(); disp[i].be = P(); }
  for (int i = 0; i < 2; i++) { dist[i].W = P(); dist[i].b = P(); dist[i].g = P(); dist[i].be = P(); }
  for (int l = 0; l < 3; l++)
    for (int br = 0; br < 2; br++)
      for (int m = 0; m < 3; m++) {
        sa[l][br][m].W = P(); sa[l][br][m].b = P();
        sa[l][br][m].g = P(); sa[l][br][m].be = P();
      }
  const float* xyz = (const float*)d_in[0];

  // ---- workspace bump allocator (deterministic layout)
  char* base = (char*)d_ws;
  size_t off = 0;
  auto allocF = [&](size_t n) { float*    p = (float*)(base + off);    off += ((n * 4 + 255) / 256) * 256; return p; };
  auto allocI = [&](size_t n) { int*      p = (int*)(base + off);      off += ((n * 4 + 255) / 256) * 256; return p; };
  auto allocH = [&](size_t n) { _Float16* p = (_Float16*)(base + off); off += ((n * 2 + 255) / 256) * 256; return p; };
  float*    pts     = allocF((size_t)B * N0 * 9);
  float*    coords0 = allocF((size_t)B * N0 * 3);
  float*    distws  = allocF((size_t)B * N0);
  int*      fpsidx  = allocI((size_t)B * 1024);
  int*      gidx    = allocI((size_t)B * 1024 * 32);
  float*    coords1 = allocF((size_t)B * 1024 * 3);
  float*    coords2 = allocF((size_t)B * 512 * 3);
  float*    coords3 = allocF((size_t)B * 256 * 3);
  float*    feats1  = allocF((size_t)B * 1024 * 64);
  float*    feats2  = allocF((size_t)B * 512 * 256);
  float*    feats3  = allocF((size_t)B * 256 * 512);
  float*    Y0      = allocF(8388608);       // f32 activations (max 65536*128)
  float*    Y1      = allocF(8388608);
  _Float16* H0      = allocH(9437184);       // f16 padded GEMM inputs (32768*288)
  _Float16* H1      = allocH(9437184);
  _Float16* l3h     = allocH(1024 * 544);
  _Float16* Wp      = allocH(147456);        // packed weights (max 256*544)
  float*    dispBuf = allocF((size_t)B * 256 * 3);
  float*    distBuf = allocF((size_t)B * 256 * 1);
  float*    bsum    = allocF(512);
  float*    bsumsq  = allocF(512);

  // pack weights + GEMM + BN(+ReLU) + f16 re-emit for next layer
  auto run_layer = [&](const _Float16* Xh, int Kp, int M, const Conv& c, int K, int O,
                       float* Y, _Float16* XhOut) {
    int Op  = (O + 63) & ~63;
    long pt = (long)Op * Kp;
    pack_w<<<(unsigned)((pt + 255) / 256), 256, 0, stream>>>(c.W, Wp, O, K, Kp, (int)pt);
    dim3 g(M / GEMM_BM, Op / GEMM_BN);
    gemm_bias_wmma<<<g, GEMM_T, 0, stream>>>(Xh, Kp, Wp, c.b, Y, M, O);
    bn_reduce<<<O, 256, 0, stream>>>(Y, M, O, bsum, bsumsq);
    int KpO = (O + 31) & ~31;
    long tot = (long)M * KpO;
    bn_apply_relu<<<(unsigned)((tot + 255) / 256), 256, 0, stream>>>(
        Y, M, O, bsum, bsumsq, c.g, c.be, XhOut, KpO);
  };

  // ---- prep
  {
    long tot = (long)B * N0 * 9;
    prep_pts<<<(unsigned)((tot + 255) / 256), 256, 0, stream>>>(xyz, pts, coords0, N0, (int)tot);
  }

  float* coordsL[4] = {coords0, coords1, coords2, coords3};
  float* featsL[4]  = {pts, feats1, feats2, feats3};
  const int   Npt[4]      = {16384, 1024, 512, 256};
  const int   featC[4]    = {9, 64, 256, 512};
  const float radii[3][2] = {{2.5f, 5.f}, {5.f, 10.f}, {10.f, 20.f}};
  const int   nsamp[2]    = {16, 32};
  const int   mlpC[3][3]  = {{9, 32, 32}, {64, 64, 128}, {256, 196, 256}};

  // ---- set-abstraction layers
  for (int l = 0; l < 3; l++) {
    int N = Npt[l], S = Npt[l + 1], Cf = featC[l];
    fps_kernel<<<B, FPS_T, 0, stream>>>(coordsL[l], N, S, fpsidx, coordsL[l + 1], distws);
    int coff = 0;
    for (int br = 0; br < 2; br++) {
      int K = nsamp[br];
      float r = radii[l][br];
      int BS = B * S;
      ball_query<<<(BS + 255) / 256, 256, 0, stream>>>(coordsL[l], coordsL[l + 1], N, S, K, r * r, gidx, BS);
      int rows = BS * K;                       // multiple of 128 for all cases
      int Cin  = Cf + 3;
      int Cinp = (Cin + 31) & ~31;
      gather_kernel<<<(rows + 255) / 256, 256, 0, stream>>>(featsL[l], Cf, coordsL[l], coordsL[l + 1],
                                                            gidx, N, S, K, H0, Cinp, rows);
      const _Float16* cur = H0; int curKp = Cinp; int curC = Cin;
      float* Yf = Y0;
      for (int m = 0; m < 3; m++) {
        Yf = (m & 1) ? Y1 : Y0;
        _Float16* hOut = (cur == H0) ? H1 : H0;
        run_layer(cur, curKp, rows, sa[l][br][m], curC, mlpC[l][m], Yf, hOut);
        cur = hOut; curKp = (mlpC[l][m] + 31) & ~31; curC = mlpC[l][m];
      }
      long pt = (long)BS * curC;
      maxpool_kernel<<<(unsigned)((pt + 255) / 256), 256, 0, stream>>>(Yf, S, K, curC,
                                                                       featsL[l + 1], featC[l + 1], coff, (int)pt);
      coff += curC;
    }
  }

  // ---- heads (M = 1024, multiple of 128)
  {
    long tot = 1024L * 544;
    concat_l3<<<(unsigned)((tot + 255) / 256), 256, 0, stream>>>(coords3, feats3, l3h, (int)tot);
  }
  run_layer(l3h, 544, 1024, disp[0], 515, 256, Y0, H0);
  run_layer(H0, 256, 1024, disp[1], 256, 3, dispBuf, H1);
  run_layer(l3h, 544, 1024, dist[0], 515, 256, Y1, H0);
  run_layer(H0, 256, 1024, dist[1], 256, 1, distBuf, H1);

  final_out<<<(7168 + 255) / 256, 256, 0, stream>>>(coords3, dispBuf, distBuf, (float*)d_out);
}
